// StandardMoE_19439021982127
// MI455X (gfx1250) — matrix operand order repaired and checked by reading, not compiled, not run on hardware
//
#include <hip/hip_runtime.h>

// ---------------------------------------------------------------------------
// MoE (top-2, 64 experts, d=1024, N=4096) for MI455X / gfx1250, wave32.
// HBM-bound (~290 MB traffic, ~12.5us floor @ 23.3 TB/s). Expert GEMMs use
// v_wmma_f32_16x16x32_bf16 (f32 accumulate) with packed fp32->bf16 staging
// through LDS; weights streamed once with prefetch.
// ---------------------------------------------------------------------------

#define N_TOKENS  4096
#define D_MODEL   1024
#define N_EXPERTS 64

#define M_TILE 64
#define N_TILE 32
#define K_TILE 64
#define LDS_STRIDE 72   // halves: 144 B/row = 9*16 -> 16B-aligned chunks, staggered banks

typedef __attribute__((ext_vector_type(16))) __bf16 v16bf;
typedef __attribute__((ext_vector_type(8)))  float  v8f;

union Frag32B { v16bf v; uint4 q[2]; };

// pack two fp32 into one dword of 2x bf16 (RNE) -> lowers to v_cvt_pk_bf16_f32
__device__ __forceinline__ unsigned int pk_bf16(float a, float b) {
    union { __bf16 h[2]; unsigned int u; } r;
    r.h[0] = (__bf16)a; r.h[1] = (__bf16)b;
    return r.u;
}

// -------------------------- init: zero out + counters ----------------------
__global__ void moe_init_kernel(float* __restrict__ out, int* __restrict__ counts) {
    int gid = blockIdx.x * blockDim.x + threadIdx.x;   // covers exactly N*D/4
    if (gid < N_EXPERTS) counts[gid] = 0;
    ((float4*)out)[gid] = make_float4(0.f, 0.f, 0.f, 0.f);
}

// -------------------------- gating: logits + top-2 -------------------------
__global__ void moe_gate_kernel(const float* __restrict__ x,
                                const float* __restrict__ gate_w,
                                int* __restrict__ counts,
                                int* __restrict__ tids,
                                float* __restrict__ tws) {
    __shared__ float lg[N_EXPERTS];
    const int n = blockIdx.x;
    const int e = threadIdx.x;

    const float4* xr = (const float4*)(x + (size_t)n * D_MODEL);
    const float4* gr = (const float4*)(gate_w + (size_t)e * D_MODEL);
    float acc = 0.f;
#pragma unroll 8
    for (int i = 0; i < D_MODEL / 4; ++i) {
        float4 a = xr[i], b = gr[i];
        acc += a.x * b.x + a.y * b.y + a.z * b.z + a.w * b.w;
    }
    lg[e] = acc;
    __syncthreads();

    if (e == 0) {
        // strict '>' ascending scan == first-occurrence tie-break (jax top_k)
        float best = -3.4e38f; int bi = 0;
        for (int j = 0; j < N_EXPERTS; ++j)
            if (lg[j] > best) { best = lg[j]; bi = j; }
        float sec = -3.4e38f; int si = 0;
        for (int j = 0; j < N_EXPERTS; ++j)
            if (j != bi && lg[j] > sec) { sec = lg[j]; si = j; }
        // softmax over the two selected logits (best >= sec)
        float t   = __expf(sec - best);
        float inv = 1.f / (1.f + t);
        int p0 = atomicAdd(&counts[bi], 1);
        tids[bi * N_TOKENS + p0] = n;  tws[bi * N_TOKENS + p0] = inv;
        int p1 = atomicAdd(&counts[si], 1);
        tids[si * N_TOKENS + p1] = n;  tws[si * N_TOKENS + p1] = t * inv;
    }
}

// ---------------- expert GEMM: gathered tokens x W_e^T, WMMA ---------------
// grid: (expert, column-tile of 32). 256 threads = 8 waves as 4(M) x 2(N).
__global__ __launch_bounds__(256)
void moe_expert_gemm_kernel(const float* __restrict__ x,
                            const float* __restrict__ expert_w,
                            const int* __restrict__ counts,
                            const int* __restrict__ tids,
                            const float* __restrict__ tws,
                            float* __restrict__ out) {
    __shared__ __align__(16) __bf16 sA[M_TILE][LDS_STRIDE];
    __shared__ __align__(16) __bf16 sB[N_TILE][LDS_STRIDE];
    __shared__ int   sTok[M_TILE];
    __shared__ float sCoef[M_TILE];

    const int e     = blockIdx.x;
    const int jbase = blockIdx.y * N_TILE;
    const int tid   = threadIdx.x;
    const int lane  = tid & 31;
    const int wave  = tid >> 5;
    const int mt    = wave & 3;     // M sub-tile (0..3)
    const int nt    = wave >> 2;    // N sub-tile (0..1)
    const int half  = lane >> 4;    // lane half selects K slice (ISA 7.12.2)
    const int l16   = lane & 15;

    const int cnt = counts[e];
    const float* wmat  = expert_w + (size_t)e * D_MODEL * D_MODEL;
    const int*   etids = tids + (size_t)e * N_TOKENS;
    const float* etws  = tws  + (size_t)e * N_TOKENS;

    // cooperative-load mappings
    const int arow = tid >> 2;            // A: 64 rows, 4 threads/row, 16 floats ea
    const int akk  = (tid & 3) << 4;
    const int bj   = tid >> 3;            // B: 32 rows, 8 threads/row, 8 floats ea
    const int bkk  = (tid & 7) << 3;

    const __bf16* sArow = &sA[mt * 16 + l16][0];
    const __bf16* sBrow = &sB[nt * 16 + l16][0];

    for (int cbase = 0; cbase < cnt; cbase += M_TILE) {
        __syncthreads();                  // previous chunk done with sTok/sCoef
        if (tid < M_TILE) {
            int idx = cbase + tid;
            if (idx < cnt) { sTok[tid] = etids[idx]; sCoef[tid] = etws[idx]; }
            else           { sTok[tid] = -1;         sCoef[tid] = 0.f; }
        }
        __syncthreads();

        // padded rows: gather x[0]; results are discarded at store time (tok<0),
        // so no zero-fill / exec-mask diamond is needed in the hot loop.
        const int tokA  = sTok[arow];
        const int gtokA = tokA < 0 ? 0 : tokA;
        const float4* asrcBase = (const float4*)(x + (size_t)gtokA * D_MODEL + akk);
        const float*  wrow     = wmat + (size_t)(jbase + bj) * D_MODEL + bkk;

        v8f acc = {0.f, 0.f, 0.f, 0.f, 0.f, 0.f, 0.f, 0.f};

        for (int k0 = 0; k0 < D_MODEL; k0 += K_TILE) {
            // ---- stage A (gathered x rows): 16 fp32 -> 2x uint4 of bf16 ----
            {
                const float4* src = asrcBase + (k0 >> 2);
                float4 f0 = src[0], f1 = src[1], f2 = src[2], f3 = src[3];
                uint4* d = (uint4*)&sA[arow][akk];
                d[0] = make_uint4(pk_bf16(f0.x, f0.y), pk_bf16(f0.z, f0.w),
                                  pk_bf16(f1.x, f1.y), pk_bf16(f1.z, f1.w));
                d[1] = make_uint4(pk_bf16(f2.x, f2.y), pk_bf16(f2.z, f2.w),
                                  pk_bf16(f3.x, f3.y), pk_bf16(f3.z, f3.w));
            }
            // ---- stage B: sB[j][k] = W_e[jbase+j][k0+k], 8 fp32 -> uint4 ----
            {
                const float4* src = (const float4*)(wrow + k0);
                __builtin_prefetch(wrow + k0 + K_TILE, 0, 1);  // next k-tile of W
                float4 w0 = src[0], w1 = src[1];
                *(uint4*)&sB[bj][bkk] =
                    make_uint4(pk_bf16(w0.x, w0.y), pk_bf16(w0.z, w0.w),
                               pk_bf16(w1.x, w1.y), pk_bf16(w1.z, w1.w));
            }
            __syncthreads();

            // ---- 2 WMMAs per staged tile (ISA 7.12.2 16-bit layouts) ----
            Frag32B fa, fb;
            fa.q[0] = *(const uint4*)&sArow[half * 8];        // K = h*8..h*8+7
            fa.q[1] = *(const uint4*)&sArow[16 + half * 8];   // K = 16+h*8..
            fb.q[0] = *(const uint4*)&sBrow[half * 8];        // K = h*8..h*8+7
            fb.q[1] = *(const uint4*)&sBrow[16 + half * 8];
            acc = __builtin_amdgcn_wmma_f32_16x16x32_bf16(
                false, fa.v, false, fb.v, (short)0, acc, false, false);

            fa.q[0] = *(const uint4*)&sArow[32 + half * 8];
            fa.q[1] = *(const uint4*)&sArow[48 + half * 8];
            fb.q[0] = *(const uint4*)&sBrow[32 + half * 8];
            fb.q[1] = *(const uint4*)&sBrow[48 + half * 8];
            acc = __builtin_amdgcn_wmma_f32_16x16x32_bf16(
                false, fa.v, false, fb.v, (short)0, acc, false, false);
            __syncthreads();
        }

        // ---- scale by routing coefficient, accumulate into out ----
        const int colg = jbase + nt * 16 + l16;     // C/D: N = lane%16
#pragma unroll
        for (int r = 0; r < 8; ++r) {               // C/D: M = r + half*8
            int m   = mt * 16 + half * 8 + r;
            int tok = sTok[m];
            if (tok >= 0)
                atomicAdd(&out[(size_t)tok * D_MODEL + colg], acc[r] * sCoef[m]);
        }
    }
}

// ---------------------------------------------------------------------------
extern "C" void kernel_launch(void* const* d_in, const int* in_sizes, int n_in,
                              void* d_out, int out_size, void* d_ws, size_t ws_size,
                              hipStream_t stream) {
    (void)in_sizes; (void)n_in; (void)out_size; (void)ws_size;
    const float* x        = (const float*)d_in[0];
    const float* gate_w   = (const float*)d_in[1];
    const float* expert_w = (const float*)d_in[2];
    float* out = (float*)d_out;

    // workspace layout (~2.1 MB): [counts 256B][tids 1MB][tws 1MB]
    char*  ws     = (char*)d_ws;
    int*   counts = (int*)ws;
    int*   tids   = (int*)(ws + 256);
    float* tws    = (float*)(ws + 256 + (size_t)N_EXPERTS * N_TOKENS * sizeof(int));

    moe_init_kernel<<<(N_TOKENS * D_MODEL / 4) / 256, 256, 0, stream>>>(out, counts);
    moe_gate_kernel<<<N_TOKENS, N_EXPERTS, 0, stream>>>(x, gate_w, counts, tids, tws);
    moe_expert_gemm_kernel<<<dim3(N_EXPERTS, D_MODEL / N_TILE), 256, 0, stream>>>(
        x, expert_w, counts, tids, tws, out);
}
